// WarpagePredictor_46377056862464
// MI455X (gfx1250) — matrix operand order repaired and checked by reading, not compiled, not run on hardware
//
#include <hip/hip_runtime.h>
#include <math.h>

#define NN 50000
#define NE 600000
#define HD 128

typedef __attribute__((ext_vector_type(16))) __bf16 v16bf;
typedef __attribute__((ext_vector_type(8)))  __bf16 v8bf;
typedef __attribute__((ext_vector_type(8)))  float  v8f;

// ---------------------------------------------------------------------------
// Core bf16 WMMA GEMM:  C[nRows,128] = A[nRows,128] (bf16) x B (bf16, stored
// transposed as BT16[col][k]).  Epilogue modes:
//   dinv != null : v = acc * dinv[row]  (GCN pre-scale; bias/relu unused)
//   else         : v = acc + bias[col]; optional relu
// Writes any of outA (f32), outB (f32), out16 (bf16).
// ---------------------------------------------------------------------------
__global__ __launch_bounds__(256)
void gemm128_wmma(const __bf16* __restrict__ A16,
                  const __bf16* __restrict__ BT16,
                  const float*  __restrict__ bias,
                  const float*  __restrict__ dinv,
                  float*        __restrict__ outA,
                  float*        __restrict__ outB,
                  __bf16*       __restrict__ out16,
                  int nRows, int relu)
{
    const int lane  = threadIdx.x & 31;
    const int wave  = threadIdx.x >> 5;
    const int row0  = blockIdx.x * 128 + wave * 16;
    if (row0 >= nRows) return;                 // wave-uniform: EXEC stays all-1s

    const int mrow  = lane & 15;               // M (A) / N (B,C) index
    const int khalf = lane >> 4;               // K half-select per ISA layout

    int arow = row0 + mrow;
    if (arow > nRows - 1) arow = nRows - 1;    // clamp; stores are guarded
    const size_t abase = (size_t)arow * HD;

    v8f acc[8] = {};

#pragma unroll
    for (int kt = 0; kt < 4; ++kt) {
        // A fragment: 16x32 bf16; lane<16 holds K = kb..kb+7 and kb+16..kb+23
        const int ka = kt * 32 + khalf * 8;
        v8bf a0 = *(const v8bf*)(A16 + abase + ka);
        v8bf a1 = *(const v8bf*)(A16 + abase + ka + 16);
        v16bf afrag;
#pragma unroll
        for (int i = 0; i < 8; ++i) { afrag[i] = a0[i]; afrag[i + 8] = a1[i]; }

        const int kb = kt * 32 + khalf * 8;    // B: same K layout as A (kb..kb+7, kb+16..kb+23)
#pragma unroll
        for (int nt = 0; nt < 8; ++nt) {
            const int col = nt * 16 + mrow;
            const __bf16* bp = BT16 + (size_t)col * HD + kb;
            v8bf b0 = *(const v8bf*)bp;
            v8bf b1 = *(const v8bf*)(bp + 16);
            v16bf bfrag;
#pragma unroll
            for (int i = 0; i < 8; ++i) { bfrag[i] = b0[i]; bfrag[i + 8] = b1[i]; }
            acc[nt] = __builtin_amdgcn_wmma_f32_16x16x32_bf16(
                false, afrag, false, bfrag, (short)0, acc[nt], false, false);
        }
    }

    // C/D layout: VGPR r of lane l holds C[row0 + r + 8*(l>>4)][nt*16 + (l&15)]
    const int rbase = khalf * 8;
    float dv[8];
#pragma unroll
    for (int r = 0; r < 8; ++r) {
        const int rw = row0 + rbase + r;
        dv[r] = (dinv && rw < nRows) ? dinv[rw] : 0.0f;
    }
#pragma unroll
    for (int nt = 0; nt < 8; ++nt) {
        const int col  = nt * 16 + mrow;
        const float bc = bias ? bias[col] : 0.0f;
#pragma unroll
        for (int r = 0; r < 8; ++r) {
            const int rw = row0 + rbase + r;
            if (rw >= nRows) continue;
            float v = acc[nt][r];
            if (dinv) {
                v *= dv[r];
            } else {
                v += bc;
                if (relu) v = fmaxf(v, 0.0f);
            }
            const size_t o = (size_t)rw * HD + col;
            if (outA)  outA[o]  = v;
            if (outB)  outB[o]  = v;
            if (out16) out16[o] = (__bf16)v;
        }
    }
}

// ---------------------------------------------------------------------------
// Layer 0 (K=8): hs = (x @ w0) * dinv, seeds agg with the self-loop term.
// 2 nodes per 256-thread block, one output dim per thread.
// ---------------------------------------------------------------------------
__global__ __launch_bounds__(256)
void gcn0_gemm(const float* __restrict__ x, const float* __restrict__ w0,
               const float* __restrict__ dinv,
               float* __restrict__ hs, float* __restrict__ agg, int nRows)
{
    const int node = blockIdx.x * 2 + (threadIdx.x >> 7);
    const int d    = threadIdx.x & 127;
    if (node >= nRows) return;
    float acc = 0.0f;
#pragma unroll
    for (int k = 0; k < 8; ++k) acc += x[node * 8 + k] * w0[k * HD + d];
    const float v = acc * dinv[node];
    const size_t o = (size_t)node * HD + d;
    hs[o]  = v;
    agg[o] = v;
}

// ---------------------------------------------------------------------------
// Degree / normalization
// ---------------------------------------------------------------------------
__global__ void deg_init(float* deg, int n)
{
    const int i = blockIdx.x * blockDim.x + threadIdx.x;
    if (i < n) deg[i] = 1.0f;   // +1 self loop
}
__global__ void deg_accum(const int* __restrict__ dst, float* deg, int nE)
{
    const int e = blockIdx.x * blockDim.x + threadIdx.x;
    if (e < nE) atomicAdd(&deg[dst[e]], 1.0f);
}
__global__ void deg_finish(float* deg, int n)
{
    const int i = blockIdx.x * blockDim.x + threadIdx.x;
    if (i < n) deg[i] = rsqrtf(deg[i]);   // in-place -> dinv
}

// ---------------------------------------------------------------------------
// Edge scatter: agg[dst] += hs[src]; 32 lanes x float4 per edge.
// ---------------------------------------------------------------------------
__global__ __launch_bounds__(256)
void edge_scatter(const int* __restrict__ src, const int* __restrict__ dst,
                  const float* __restrict__ hs, float* __restrict__ agg, int nE)
{
    const long long idx = (long long)blockIdx.x * 256 + threadIdx.x;
    const int e = (int)(idx >> 5);
    if (e >= nE) return;
    const int d0 = (int)(idx & 31) * 4;
    const int s = src[e], t = dst[e];
    const float4 m = *(const float4*)(hs + (size_t)s * HD + d0);
    float* ap = agg + (size_t)t * HD + d0;
    atomicAdd(ap + 0, m.x);
    atomicAdd(ap + 1, m.y);
    atomicAdd(ap + 2, m.z);
    atomicAdd(ap + 3, m.w);
}

// ---------------------------------------------------------------------------
// GCN epilogue: out = agg*dinv + b (optional relu), emitted as bf16 for the
// next GEMM's A operand.
// ---------------------------------------------------------------------------
__global__ __launch_bounds__(256)
void finalize_gcn(const float* __restrict__ agg, const float* __restrict__ dinv,
                  const float* __restrict__ bias, __bf16* __restrict__ out16,
                  int nRows, int relu)
{
    const int idx = blockIdx.x * blockDim.x + threadIdx.x;
    if (idx >= nRows * HD) return;
    const int n = idx >> 7, d = idx & 127;
    float v = agg[idx] * dinv[n] + bias[d];
    if (relu) v = fmaxf(v, 0.0f);
    out16[idx] = (__bf16)v;
}

// ---------------------------------------------------------------------------
// Weight prep: fp32 [K,N] -> bf16 transposed [N,K]
// ---------------------------------------------------------------------------
__global__ void w_to_bf16_T(const float* __restrict__ W, __bf16* __restrict__ WT,
                            int K, int N)
{
    const int idx = blockIdx.x * blockDim.x + threadIdx.x;
    if (idx >= K * N) return;
    const int k = idx / N, n = idx % N;
    WT[n * K + k] = (__bf16)W[idx];
}

// Attention folding: softmax over a single key == 1 => att = h @ (Wv@Wo) + (bv@Wo + bo)
__global__ void build_weff(const float* __restrict__ inW,   // [128,384]
                           const float* __restrict__ outW,  // [128,128]
                           __bf16* __restrict__ weffT)      // [n][k]
{
    const int idx = blockIdx.x * blockDim.x + threadIdx.x;
    if (idx >= HD * HD) return;
    const int n = idx >> 7, k = idx & 127;
    float acc = 0.0f;
    for (int j = 0; j < HD; ++j)
        acc += inW[k * 384 + 2 * HD + j] * outW[j * HD + n];
    weffT[n * HD + k] = (__bf16)acc;
}
__global__ void build_beff(const float* __restrict__ inB, const float* __restrict__ outW,
                           const float* __restrict__ outB, float* __restrict__ beff)
{
    const int n = threadIdx.x;
    if (n >= HD) return;
    float acc = outB[n];
    for (int j = 0; j < HD; ++j) acc += inB[2 * HD + j] * outW[j * HD + n];
    beff[n] = acc;
}

// Fuse fc0 (128->64) and vd0 (128->64) into one 128->128 GEMM.
__global__ void build_wheads(const float* __restrict__ fc0w, const float* __restrict__ fc0b,
                             const float* __restrict__ vd0w, const float* __restrict__ vd0b,
                             __bf16* __restrict__ whT, float* __restrict__ bh)
{
    const int idx = blockIdx.x * blockDim.x + threadIdx.x;
    if (idx >= HD * HD) return;
    const int n = idx / HD, k = idx % HD;
    const float v = (n < 64) ? fc0w[k * 64 + n] : vd0w[k * 64 + (n - 64)];
    whT[n * HD + k] = (__bf16)v;
    if (k == 0) bh[n] = (n < 64) ? fc0b[n] : vd0b[n - 64];
}

// ---------------------------------------------------------------------------
// Head tails: u[:,0:64] -> fc1(relu) -> fc2 -> wp[3]; u[:,64:128] -> vd1 -> sigmoid
// ---------------------------------------------------------------------------
__global__ __launch_bounds__(256)
void heads_tail(const float* __restrict__ u,
                const float* __restrict__ fc1w, const float* __restrict__ fc1b,
                const float* __restrict__ fc2w, const float* __restrict__ fc2b,
                const float* __restrict__ vd1w, const float* __restrict__ vd1b,
                float* __restrict__ out, int nRows)
{
    const int n = blockIdx.x * blockDim.x + threadIdx.x;
    if (n >= nRows) return;
    const float* ur = u + (size_t)n * HD;

    float acc[32];
#pragma unroll
    for (int j = 0; j < 32; ++j) acc[j] = fc1b[j];
#pragma unroll 4
    for (int k = 0; k < 64; ++k) {
        const float a = ur[k];
#pragma unroll
        for (int j = 0; j < 32; ++j) acc[j] += a * fc1w[k * 32 + j];
    }
    float wp0 = fc2b[0], wp1 = fc2b[1], wp2 = fc2b[2];
#pragma unroll
    for (int j = 0; j < 32; ++j) {
        const float a = fmaxf(acc[j], 0.0f);
        wp0 += a * fc2w[j * 3 + 0];
        wp1 += a * fc2w[j * 3 + 1];
        wp2 += a * fc2w[j * 3 + 2];
    }
    out[(size_t)n * 3 + 0] = wp0;
    out[(size_t)n * 3 + 1] = wp1;
    out[(size_t)n * 3 + 2] = wp2;

    float v = vd1b[0];
#pragma unroll 4
    for (int k = 0; k < 64; ++k) v += ur[64 + k] * vd1w[k];
    out[(size_t)nRows * 3 + n] = 1.0f / (1.0f + expf(-v));
}

// ---------------------------------------------------------------------------
extern "C" void kernel_launch(void* const* d_in, const int* in_sizes, int n_in,
                              void* d_out, int out_size, void* d_ws, size_t ws_size,
                              hipStream_t stream)
{
    const float* x       = (const float*)d_in[0];
    const int*   eidx    = (const int*)d_in[1];
    const float* gcn_w[4] = {(const float*)d_in[2], (const float*)d_in[4],
                             (const float*)d_in[6], (const float*)d_in[8]};
    const float* gcn_b[4] = {(const float*)d_in[3], (const float*)d_in[5],
                             (const float*)d_in[7], (const float*)d_in[9]};
    const float* attn_in_w  = (const float*)d_in[10];
    const float* attn_in_b  = (const float*)d_in[11];
    const float* attn_out_w = (const float*)d_in[12];
    const float* attn_out_b = (const float*)d_in[13];
    const float* fc0w = (const float*)d_in[14], *fc0b = (const float*)d_in[15];
    const float* fc1w = (const float*)d_in[16], *fc1b = (const float*)d_in[17];
    const float* fc2w = (const float*)d_in[18], *fc2b = (const float*)d_in[19];
    const float* vd0w = (const float*)d_in[20], *vd0b = (const float*)d_in[21];
    const float* vd1w = (const float*)d_in[22], *vd1b = (const float*)d_in[23];
    float* out = (float*)d_out;

    const int* srcI = eidx;
    const int* dstI = eidx + NE;

    // ---- workspace carve-up (256B aligned) ----
    char* ws = (char*)d_ws;
    size_t off = 0;
    auto carve = [&](size_t bytes) -> void* {
        off = (off + 255) & ~(size_t)255;
        void* p = ws + off;
        off += bytes;
        return p;
    };
    float*  bufHS  = (float*)carve((size_t)NN * HD * 4);   // hs (also reused as u)
    float*  bufAGG = (float*)carve((size_t)NN * HD * 4);   // atomic aggregation
    __bf16* bufX16 = (__bf16*)carve((size_t)NN * HD * 2);  // activations bf16
    __bf16* bufY16 = (__bf16*)carve((size_t)NN * HD * 2);  // attention out bf16
    float*  dinv   = (float*)carve((size_t)NN * 4);
    __bf16* wbT[3];
    for (int i = 0; i < 3; ++i) wbT[i] = (__bf16*)carve((size_t)HD * HD * 2);
    __bf16* weffT  = (__bf16*)carve((size_t)HD * HD * 2);
    __bf16* whT    = (__bf16*)carve((size_t)HD * HD * 2);
    float*  beff   = (float*)carve(HD * 4);
    float*  bh     = (float*)carve(HD * 4);
    (void)ws_size; (void)n_in; (void)in_sizes; (void)out_size;

    const int gemmGrid = (NN + 127) / 128;
    const int elemGrid = (NN * HD + 255) / 256;

    // ---- one-time (per call) weight prep ----
    for (int i = 0; i < 3; ++i)
        w_to_bf16_T<<<(HD * HD + 255) / 256, 256, 0, stream>>>(gcn_w[i + 1], wbT[i], HD, HD);
    build_weff<<<(HD * HD + 255) / 256, 256, 0, stream>>>(attn_in_w, attn_out_w, weffT);
    build_beff<<<1, HD, 0, stream>>>(attn_in_b, attn_out_w, attn_out_b, beff);
    build_wheads<<<(HD * HD + 255) / 256, 256, 0, stream>>>(fc0w, fc0b, vd0w, vd0b, whT, bh);

    // ---- degree / dinv ----
    deg_init<<<(NN + 255) / 256, 256, 0, stream>>>(dinv, NN);
    deg_accum<<<(NE + 255) / 256, 256, 0, stream>>>(dstI, dinv, NE);
    deg_finish<<<(NN + 255) / 256, 256, 0, stream>>>(dinv, NN);

    // ---- GCN layer 0 (K=8, VALU) ----
    gcn0_gemm<<<(NN + 1) / 2, 256, 0, stream>>>(x, gcn_w[0], dinv, bufHS, bufAGG, NN);
    edge_scatter<<<(int)(((long long)NE * 32 + 255) / 256), 256, 0, stream>>>(
        srcI, dstI, bufHS, bufAGG, NE);
    finalize_gcn<<<elemGrid, 256, 0, stream>>>(bufAGG, dinv, gcn_b[0], bufX16, NN, 1);

    // ---- GCN layers 1..3 (bf16 WMMA GEMM + scatter) ----
    for (int l = 1; l <= 3; ++l) {
        gemm128_wmma<<<gemmGrid, 256, 0, stream>>>(
            bufX16, wbT[l - 1], nullptr, dinv, bufHS, bufAGG, nullptr, NN, 0);
        edge_scatter<<<(int)(((long long)NE * 32 + 255) / 256), 256, 0, stream>>>(
            srcI, dstI, bufHS, bufAGG, NE);
        finalize_gcn<<<elemGrid, 256, 0, stream>>>(
            bufAGG, dinv, gcn_b[l], bufX16, NN, l < 3 ? 1 : 0);
    }

    // ---- folded attention: att = h @ (Wv@Wo) + beff ----
    gemm128_wmma<<<gemmGrid, 256, 0, stream>>>(
        bufX16, weffT, beff, nullptr, nullptr, nullptr, bufY16, NN, 0);

    // ---- fused fc0|vd0 (relu) ----
    gemm128_wmma<<<gemmGrid, 256, 0, stream>>>(
        bufY16, whT, bh, nullptr, bufHS, nullptr, nullptr, NN, 1);

    // ---- head tails ----
    heads_tail<<<(NN + 255) / 256, 256, 0, stream>>>(
        bufHS, fc1w, fc1b, fc2w, fc2b, vd1w, vd1b, out, NN);
}
